// PrunableMixtralSparseMoeBlockWrapper_15023795601822
// MI455X (gfx1250) — matrix operand order repaired and checked by reading, not compiled, not run on hardware
//
#include <hip/hip_runtime.h>
#include <hip/hip_bf16.h>
#include <cstddef>

// ---------------------------------------------------------------------------
// Mixtral sparse MoE block for MI455X (gfx1250, wave32, WMMA).
// B=2,S=2048 -> T=4096 tokens, H=1024, FF=2048, E=8, K=2.
// Sparse (gathered) expert compute == reference dense-combine math.
// bf16 WMMA 16x16x32, fp32 accumulation. Weights/activations pre-converted
// to bf16 in workspace (fallback converts in-flight if ws is too small).
//
// PRE path LDS = 132 KB -> 2 workgroups/WGP (4 waves/SIMD) for latency hiding;
// A fragments stream from L2-resident bf16 copies, B fragments from L2-resident
// bf16 weights; each A pair feeds 8 WMMAs (two N-chunks per wave iteration).
// ---------------------------------------------------------------------------

#define T_TOK  4096
#define H_DIM  1024
#define FF_DIM 2048
#define NEXP   8
#define MTILE  32            // tokens per expert tile (2 WMMA M-tiles)

#define SX_STRIDE (H_DIM + 8)     // bf16 elements per LDS x-row (!PRE only)
#define SI_STRIDE (FF_DIM + 8)    // bf16 elements per LDS inner-row
#define SX_BYTES  (2 * MTILE * SX_STRIDE)
#define SI_BYTES  (2 * MTILE * SI_STRIDE)
#define SMEM_PRE   (256 + SI_BYTES)              // ~132 KB
#define SMEM_NOPRE (256 + SX_BYTES + SI_BYTES)   // ~198 KB

typedef __attribute__((ext_vector_type(16))) __bf16 v16bf;
typedef __attribute__((ext_vector_type(8)))  __bf16 v8bf;
typedef __attribute__((ext_vector_type(4)))  __bf16 v4bf;
typedef __attribute__((ext_vector_type(8)))  float  v8f;

// A-matrix (16-bit, 16x32) fragment, source row-major with K contiguous.
// p must already include the per-lane +16 offset for lanes 16..31:
// lane<16:  K = +0..15 ; lane>=16: K = +16..31.
__device__ __forceinline__ v16bf load_frag_gap16(const __bf16* p) {
    v8bf lo = *(const v8bf*)(p);
    v8bf hi = *(const v8bf*)(p + 8);
    return __builtin_shufflevector(lo, hi, 0,1,2,3,4,5,6,7,8,9,10,11,12,13,14,15);
}

// B-matrix (16-bit, 32x16) fragment: lane holds 16 contiguous K values of
// column n = lane%16 (K base += 16 for lanes 16..31). bf16 source.
__device__ __forceinline__ v16bf load_b_bf16(const __bf16* p) {
    v8bf lo = *(const v8bf*)(p);
    v8bf hi = *(const v8bf*)(p + 8);
    return __builtin_shufflevector(lo, hi, 0,1,2,3,4,5,6,7,8,9,10,11,12,13,14,15);
}

// Fallback: B fragment from fp32 source, converted in-flight.
__device__ __forceinline__ v16bf load_b_f32(const float* p) {
    v16bf b;
#pragma unroll
    for (int i = 0; i < 4; ++i) {
        float4 v = ((const float4*)p)[i];
        b[4*i+0] = (__bf16)v.x;
        b[4*i+1] = (__bf16)v.y;
        b[4*i+2] = (__bf16)v.z;
        b[4*i+3] = (__bf16)v.w;
    }
    return b;
}

__device__ __forceinline__ float fast_silu(float x) {
    return x * __builtin_amdgcn_rcpf(1.f + __expf(-x));   // single v_rcp_f32
}

__device__ __forceinline__ v8f wmma_bf16(v16bf a, v16bf b, v8f c) {
    return __builtin_amdgcn_wmma_f32_16x16x32_bf16(false, a, false, b,
                                                   (short)0, c, false, false);
}

// ---------------------------------------------------------------------------
// Kernel 0: zero final-output accumulator region.
// ---------------------------------------------------------------------------
__global__ __launch_bounds__(256) void moe_zero_kernel(float* __restrict__ p, int n4) {
    int i = blockIdx.x * blockDim.x + threadIdx.x;
    if (i < n4) ((float4*)p)[i] = make_float4(0.f, 0.f, 0.f, 0.f);
}

// ---------------------------------------------------------------------------
// Kernel 0b: fp32 -> bf16 bulk conversion (x and weights, once per call).
// ---------------------------------------------------------------------------
__global__ __launch_bounds__(256)
void moe_cvt_bf16_kernel(const float* __restrict__ src, __bf16* __restrict__ dst, int n4) {
    int i = blockIdx.x * blockDim.x + threadIdx.x;
    if (i >= n4) return;
    float4 v = ((const float4*)src)[i];
    v4bf o;
    o[0] = (__bf16)v.x; o[1] = (__bf16)v.y; o[2] = (__bf16)v.z; o[3] = (__bf16)v.w;
    ((v4bf*)dst)[i] = o;
}

// ---------------------------------------------------------------------------
// Kernel 1: router. One wave per token. Writes raw fp32 logits (output #2),
// plus top-2 expert ids and renormalized weights.
// ---------------------------------------------------------------------------
__global__ __launch_bounds__(256)
void moe_router_kernel(const float* __restrict__ x, const float* __restrict__ gw,
                       float* __restrict__ logits_out,
                       int* __restrict__ sel, float* __restrict__ selw) {
    const int t    = blockIdx.x * 8 + (threadIdx.x >> 5);
    const int lane = threadIdx.x & 31;

    float acc[NEXP];
#pragma unroll
    for (int e = 0; e < NEXP; ++e) acc[e] = 0.f;

    const float* xt = x + (size_t)t * H_DIM;
    for (int h = lane; h < H_DIM; h += 32) {
        float xv = xt[h];
#pragma unroll
        for (int e = 0; e < NEXP; ++e) acc[e] += xv * gw[e * H_DIM + h];
    }
#pragma unroll
    for (int e = 0; e < NEXP; ++e) {
#pragma unroll
        for (int off = 16; off > 0; off >>= 1)
            acc[e] += __shfl_xor(acc[e], off, 32);
    }

    if (lane == 0) {
        float mx = acc[0];
#pragma unroll
        for (int e = 1; e < NEXP; ++e) mx = fmaxf(mx, acc[e]);
        float p[NEXP];
#pragma unroll
        for (int e = 0; e < NEXP; ++e) p[e] = __expf(acc[e] - mx);
        int a0 = 0;                       // ties -> lowest index (lax.top_k)
#pragma unroll
        for (int e = 1; e < NEXP; ++e) if (p[e] > p[a0]) a0 = e;
        int a1 = (a0 == 0) ? 1 : 0;
#pragma unroll
        for (int e = 0; e < NEXP; ++e) if (e != a0 && p[e] > p[a1]) a1 = e;
        float inv = __builtin_amdgcn_rcpf(p[a0] + p[a1]);  // softmax denom cancels
#pragma unroll
        for (int e = 0; e < NEXP; ++e) logits_out[(size_t)t * NEXP + e] = acc[e];
        sel[t * 2 + 0]  = a0;
        sel[t * 2 + 1]  = a1;
        selw[t * 2 + 0] = p[a0] * inv;
        selw[t * 2 + 1] = p[a1] * inv;
    }
}

// ---------------------------------------------------------------------------
// Kernel 2: deterministic per-expert gather lists via LDS prefix scan.
// ---------------------------------------------------------------------------
__global__ __launch_bounds__(256)
void moe_build_lists_kernel(const int* __restrict__ sel, const float* __restrict__ selw,
                            int* __restrict__ counts, int* __restrict__ tokenlist,
                            float* __restrict__ weightlist) {
    __shared__ int sCnt[256];
    const int e   = blockIdx.x;
    const int tid = threadIdx.x;
    const int t0  = tid * (T_TOK / 256);

    int   ltok[T_TOK / 256];
    float lw[T_TOK / 256];
    int c = 0;
    for (int i = 0; i < T_TOK / 256; ++i) {
        int t  = t0 + i;
        int s0 = sel[t * 2 + 0];
        int s1 = sel[t * 2 + 1];
        if (s0 == e)      { ltok[c] = t; lw[c] = selw[t * 2 + 0]; ++c; }
        else if (s1 == e) { ltok[c] = t; lw[c] = selw[t * 2 + 1]; ++c; }
    }
    sCnt[tid] = c;
    __syncthreads();
    for (int off = 1; off < 256; off <<= 1) {
        int v = (tid >= off) ? sCnt[tid - off] : 0;
        __syncthreads();
        sCnt[tid] += v;
        __syncthreads();
    }
    int base = sCnt[tid] - c;
    for (int i = 0; i < c; ++i) {
        tokenlist[(size_t)e * T_TOK + base + i]  = ltok[i];
        weightlist[(size_t)e * T_TOK + base + i] = lw[i];
    }
    if (tid == 255) counts[e] = sCnt[255];
}

// ---------------------------------------------------------------------------
// Kernel 3: fused expert SwiGLU MLP over a gathered 32-token tile.
// grid = (E, T/32); blocks past counts[e] exit immediately.
// 256 threads = 8 waves; each wave owns TWO 16-wide N slices x two M tiles
// (8 WMMAs per A-fragment pair in the up-projection).
// ---------------------------------------------------------------------------
template <bool PRE>
__global__ __launch_bounds__(256)
void moe_expert_mlp_kernel(const float*  __restrict__ x,
                           const __bf16* __restrict__ xb,
                           const float*  __restrict__ w1f,
                           const float*  __restrict__ w2f,
                           const float*  __restrict__ w3f,
                           const __bf16* __restrict__ w1b,
                           const __bf16* __restrict__ w2b,
                           const __bf16* __restrict__ w3b,
                           const int*    __restrict__ counts,
                           const int*    __restrict__ tokenlist,
                           const float*  __restrict__ weightlist,
                           float*        __restrict__ out) {
    extern __shared__ __align__(16) char smem[];
    int*    sTok   = (int*)smem;
    float*  sWgt   = (float*)(smem + 128);
    __bf16* sX     = (__bf16*)(smem + 256);                          // !PRE only
    __bf16* sInner = (__bf16*)(smem + 256 + (PRE ? 0 : SX_BYTES));

    const int e    = blockIdx.x;
    const int tile = blockIdx.y;
    const int cnt  = counts[e];
    if (tile * MTILE >= cnt) return;

    const int tid  = threadIdx.x;
    const int lane = tid & 31;
    const int wv   = tid >> 5;

    if (tid < MTILE) {
        int idx = tile * MTILE + tid;
        if (idx < cnt) {
            sTok[tid] = tokenlist[(size_t)e * T_TOK + idx];
            sWgt[tid] = weightlist[(size_t)e * T_TOK + idx];
        } else {
            sTok[tid] = 0;        // padded row: weight 0 -> contributes nothing
            sWgt[tid] = 0.f;
        }
    }
    __syncthreads();

    // ---- Phase A (!PRE only): gather + convert x rows into LDS ----
    if constexpr (!PRE) {
        const int row  = tid >> 3;            // 0..31
        const int col0 = (tid & 7) * 128;     // 128 elements per thread
        __bf16* dst = sX + (size_t)row * SX_STRIDE + col0;
        const float4* src = (const float4*)(x + (size_t)sTok[row] * H_DIM + col0);
#pragma unroll
        for (int i = 0; i < 32; ++i) {
            float4 v = src[i];
            dst[4*i+0] = (__bf16)v.x; dst[4*i+1] = (__bf16)v.y;
            dst[4*i+2] = (__bf16)v.z; dst[4*i+3] = (__bf16)v.w;
        }
        __syncthreads();
    }

    const int m     = lane & 15;
    const int nl    = lane & 15;
    const int aoff  = (lane & 16) ? 16 : 0;   // A-fragment per-lane K offset
    const int koff  = (lane >> 4) << 4;       // B-fragment per-lane K offset
    const int mbase = (lane & 16) ? 8 : 0;

    // PRE path: A fragments stream straight from L2-resident bf16 x copy.
    const __bf16* pa0 = nullptr;
    const __bf16* pa1 = nullptr;
    if constexpr (PRE) {
        pa0 = xb + (size_t)sTok[m]      * H_DIM + aoff;
        pa1 = xb + (size_t)sTok[m + 16] * H_DIM + aoff;
    }

    // ---- Phase B: up-projections + SwiGLU (two N-chunks per iteration) ----
    const size_t eW   = (size_t)e * FF_DIM * H_DIM;
    const size_t ROW16 = (size_t)16 * H_DIM;           // +16 FF rows
    for (int ff0 = wv * 32; ff0 < FF_DIM; ff0 += 8 * 32) {
        v8f acc1[2][2] = {};    // [n-chunk][m-tile]
        v8f acc3[2][2] = {};
        const __bf16* pb1 = nullptr; const __bf16* pb3 = nullptr;
        const float*  qb1 = nullptr; const float*  qb3 = nullptr;
        if constexpr (PRE) {
            pb1 = w1b + eW + (size_t)(ff0 + nl) * H_DIM + koff;
            pb3 = w3b + eW + (size_t)(ff0 + nl) * H_DIM + koff;
            // keep next iteration's weight rows flowing toward the WGP
            __builtin_prefetch(pb1 + 8 * 32 * (size_t)H_DIM, 0, 1);
            __builtin_prefetch(pb3 + 8 * 32 * (size_t)H_DIM, 0, 1);
        } else {
            qb1 = w1f + eW + (size_t)(ff0 + nl) * H_DIM + koff;
            qb3 = w3f + eW + (size_t)(ff0 + nl) * H_DIM + koff;
        }
        for (int kt = 0; kt < H_DIM; kt += 32) {
            v16bf a0, a1;
            if constexpr (PRE) {
                a0 = load_frag_gap16(pa0 + kt);
                a1 = load_frag_gap16(pa1 + kt);
            } else {
                a0 = load_frag_gap16(sX + (size_t)m        * SX_STRIDE + aoff + kt);
                a1 = load_frag_gap16(sX + (size_t)(m + 16) * SX_STRIDE + aoff + kt);
            }
            v16bf b1c0, b1c1, b3c0, b3c1;
            if constexpr (PRE) {
                b1c0 = load_b_bf16(pb1 + kt);  b1c1 = load_b_bf16(pb1 + ROW16 + kt);
                b3c0 = load_b_bf16(pb3 + kt);  b3c1 = load_b_bf16(pb3 + ROW16 + kt);
            } else {
                b1c0 = load_b_f32(qb1 + kt);   b1c1 = load_b_f32(qb1 + ROW16 + kt);
                b3c0 = load_b_f32(qb3 + kt);   b3c1 = load_b_f32(qb3 + ROW16 + kt);
            }
            acc1[0][0] = wmma_bf16(a0, b1c0, acc1[0][0]);
            acc1[0][1] = wmma_bf16(a1, b1c0, acc1[0][1]);
            acc1[1][0] = wmma_bf16(a0, b1c1, acc1[1][0]);
            acc1[1][1] = wmma_bf16(a1, b1c1, acc1[1][1]);
            acc3[0][0] = wmma_bf16(a0, b3c0, acc3[0][0]);
            acc3[0][1] = wmma_bf16(a1, b3c0, acc3[0][1]);
            acc3[1][0] = wmma_bf16(a0, b3c1, acc3[1][0]);
            acc3[1][1] = wmma_bf16(a1, b3c1, acc3[1][1]);
        }
#pragma unroll
        for (int c = 0; c < 2; ++c) {
            const int nOut = ff0 + c * 16 + nl;
#pragma unroll
            for (int r = 0; r < 8; ++r) {
                sInner[(size_t)(mbase + r)      * SI_STRIDE + nOut] =
                    (__bf16)(fast_silu(acc1[c][0][r]) * acc3[c][0][r]);
                sInner[(size_t)(16 + mbase + r) * SI_STRIDE + nOut] =
                    (__bf16)(fast_silu(acc1[c][1][r]) * acc3[c][1][r]);
            }
        }
    }
    __syncthreads();

    // ---- Phase C: down-projection + weighted scatter-add ----
    const size_t eW2   = (size_t)e * H_DIM * FF_DIM;
    const size_t ROW16C = (size_t)16 * FF_DIM;
    for (int n0 = wv * 32; n0 < H_DIM; n0 += 8 * 32) {
        v8f acc[2][2] = {};     // [n-chunk][m-tile]
        const __bf16* pbb = nullptr; const float* qbb = nullptr;
        if constexpr (PRE) {
            pbb = w2b + eW2 + (size_t)(n0 + nl) * FF_DIM + koff;
            __builtin_prefetch(pbb + 8 * 32 * (size_t)FF_DIM, 0, 1);
        } else {
            qbb = w2f + eW2 + (size_t)(n0 + nl) * FF_DIM + koff;
        }
        for (int kt = 0; kt < FF_DIM; kt += 32) {
            v16bf a0 = load_frag_gap16(sInner + (size_t)m        * SI_STRIDE + aoff + kt);
            v16bf a1 = load_frag_gap16(sInner + (size_t)(m + 16) * SI_STRIDE + aoff + kt);
            v16bf bc0, bc1;
            if constexpr (PRE) {
                bc0 = load_b_bf16(pbb + kt);
                bc1 = load_b_bf16(pbb + ROW16C + kt);
            } else {
                bc0 = load_b_f32(qbb + kt);
                bc1 = load_b_f32(qbb + ROW16C + kt);
            }
            acc[0][0] = wmma_bf16(a0, bc0, acc[0][0]);
            acc[0][1] = wmma_bf16(a1, bc0, acc[0][1]);
            acc[1][0] = wmma_bf16(a0, bc1, acc[1][0]);
            acc[1][1] = wmma_bf16(a1, bc1, acc[1][1]);
        }
#pragma unroll
        for (int c = 0; c < 2; ++c) {
            const int nc = n0 + c * 16 + nl;
#pragma unroll
            for (int r = 0; r < 8; ++r) {
                int mm0 = mbase + r;
                int mm1 = 16 + mbase + r;
                atomicAdd(&out[(size_t)sTok[mm0] * H_DIM + nc], acc[c][0][r] * sWgt[mm0]);
                atomicAdd(&out[(size_t)sTok[mm1] * H_DIM + nc], acc[c][1][r] * sWgt[mm1]);
            }
        }
    }
}

// ---------------------------------------------------------------------------
// Host launcher
// ---------------------------------------------------------------------------
extern "C" void kernel_launch(void* const* d_in, const int* in_sizes, int n_in,
                              void* d_out, int out_size, void* d_ws, size_t ws_size,
                              hipStream_t stream) {
    (void)in_sizes; (void)n_in; (void)out_size;

    const float* x  = (const float*)d_in[0];   // [T, H]
    const float* gw = (const float*)d_in[1];   // [E, H]
    const float* w1 = (const float*)d_in[2];   // [E, FF, H]
    const float* w2 = (const float*)d_in[3];   // [E, H, FF]
    const float* w3 = (const float*)d_in[4];   // [E, FF, H]

    float* final_out = (float*)d_out;                       // [T, H]
    float* logits    = final_out + (size_t)T_TOK * H_DIM;   // [T, E]

    // Workspace layout
    const size_t W_ELEMS  = (size_t)NEXP * FF_DIM * H_DIM;  // per weight tensor
    const size_t X_ELEMS  = (size_t)T_TOK * H_DIM;
    const size_t OFF_XB   = 512 * 1024;                     // after list region
    const size_t OFF_W1B  = OFF_XB  + X_ELEMS * 2;
    const size_t OFF_W2B  = OFF_W1B + W_ELEMS * 2;
    const size_t OFF_W3B  = OFF_W2B + W_ELEMS * 2;
    const size_t WS_NEED  = OFF_W3B + W_ELEMS * 2;          // ~104.5 MB

    char*  ws         = (char*)d_ws;
    int*   counts     = (int*)ws;
    int*   sel        = (int*)(ws + 256);
    float* selw       = (float*)(ws + 256 + sizeof(int) * 2 * T_TOK);
    int*   tokenlist  = (int*)(ws + 256 + 8 * 2 * T_TOK);
    float* weightlist = (float*)(ws + 256 + 8 * 2 * T_TOK + sizeof(int) * NEXP * T_TOK);
    __bf16* xb  = (__bf16*)(ws + OFF_XB);
    __bf16* w1b = (__bf16*)(ws + OFF_W1B);
    __bf16* w2b = (__bf16*)(ws + OFF_W2B);
    __bf16* w3b = (__bf16*)(ws + OFF_W3B);

    const bool pre = (ws_size >= WS_NEED);   // fixed per run -> deterministic

    // 0) zero the final-output accumulator (harness poisons d_out)
    {
        int n4 = (T_TOK * H_DIM) / 4;
        moe_zero_kernel<<<(n4 + 255) / 256, 256, 0, stream>>>(final_out, n4);
    }
    // 0b) bf16 staging of x and weights
    if (pre) {
        int xn4 = (int)(X_ELEMS / 4);
        int wn4 = (int)(W_ELEMS / 4);
        moe_cvt_bf16_kernel<<<(xn4 + 255) / 256, 256, 0, stream>>>(x,  xb,  xn4);
        moe_cvt_bf16_kernel<<<(wn4 + 255) / 256, 256, 0, stream>>>(w1, w1b, wn4);
        moe_cvt_bf16_kernel<<<(wn4 + 255) / 256, 256, 0, stream>>>(w2, w2b, wn4);
        moe_cvt_bf16_kernel<<<(wn4 + 255) / 256, 256, 0, stream>>>(w3, w3b, wn4);
    }
    // 1) router
    moe_router_kernel<<<T_TOK / 8, 256, 0, stream>>>(x, gw, logits, sel, selw);
    // 2) deterministic gather-list build
    moe_build_lists_kernel<<<NEXP, 256, 0, stream>>>(sel, selw, counts, tokenlist,
                                                     weightlist);
    // 3) fused expert MLPs
    dim3 grid(NEXP, T_TOK / MTILE);
    if (pre) {
        moe_expert_mlp_kernel<true><<<grid, 256, SMEM_PRE, stream>>>(
            x, xb, w1, w2, w3, w1b, w2b, w3b, counts, tokenlist, weightlist, final_out);
    } else {
        moe_expert_mlp_kernel<false><<<grid, 256, SMEM_NOPRE, stream>>>(
            x, nullptr, w1, w2, w3, nullptr, nullptr, nullptr,
            counts, tokenlist, weightlist, final_out);
    }
}
